// AtenMatmulQMixedSigni8_54752243089869
// MI455X (gfx1250) — 8-bit matrix operand order repaired and checked by reading, compiled, not run on hardware
//
#include <hip/hip_runtime.h>
#include <stdint.h>

typedef __attribute__((ext_vector_type(8))) int v8i;
typedef int v4i_vs __attribute__((vector_size(16)));

#define S_DIM 7
#define B_DIM 8
#define MDIM 1024
#define NDIM 1024
#define KDIM 1024

#if defined(__has_builtin)
#if __has_builtin(__builtin_amdgcn_global_load_async_to_lds_b128)
#define HAVE_ASYNC_LDS 1
#endif
#endif

__device__ __forceinline__ void async_wait_all() {
#if defined(__has_builtin) && __has_builtin(__builtin_amdgcn_s_wait_asynccnt)
    __builtin_amdgcn_s_wait_asynccnt(0);
#else
    asm volatile("s_wait_asynccnt 0x0" ::: "memory");
#endif
}

__device__ __forceinline__ void copy16_g2l(const uint8_t* __restrict__ g, uint8_t* l) {
#ifdef HAVE_ASYNC_LDS
    __builtin_amdgcn_global_load_async_to_lds_b128(
        (__attribute__((address_space(1))) v4i_vs*)(uintptr_t)g,
        (__attribute__((address_space(3))) v4i_vs*)(uint32_t)(uintptr_t)l,
        0, 0);
#else
    *reinterpret_cast<uint4*>(l) = *reinterpret_cast<const uint4*>(g);
#endif
}

__global__ __launch_bounds__(256) void conv_x_kernel(
    const float* __restrict__ X, uint32_t* __restrict__ Xq, int* __restrict__ Rs)
{
    const int row = blockIdx.x;
    const int tid = threadIdx.x;

    const float4 f = *reinterpret_cast<const float4*>(X + (size_t)row * KDIM + tid * 4);
    const int i0 = (int)f.x, i1 = (int)f.y, i2 = (int)f.z, i3 = (int)f.w;
    Xq[(size_t)row * (KDIM / 4) + tid] =
        ((uint32_t)(i0 & 255))       | ((uint32_t)(i1 & 255) << 8) |
        ((uint32_t)(i2 & 255) << 16) | ((uint32_t)(i3 & 255) << 24);

    int part = i0 + i1 + i2 + i3;
#pragma unroll
    for (int off = 16; off > 0; off >>= 1) part += __shfl_down(part, off, 32);

    __shared__ int wred[8];
    if ((tid & 31) == 0) wred[tid >> 5] = part;
    __syncthreads();
    if (tid == 0) {
        int s = 0;
#pragma unroll
        for (int i = 0; i < 8; ++i) s += wred[i];
        Rs[row] = s;
    }
}

__global__ __launch_bounds__(256) void conv_y_kernel(
    const float* __restrict__ Y, uint8_t* __restrict__ Ytq, int* __restrict__ Cs)
{
    __shared__ __attribute__((aligned(16))) uint8_t t[64 * 80];

    const int tid = threadIdx.x;
    const int b   = blockIdx.z;
    const int k0  = blockIdx.y * 64;
    const int n0  = blockIdx.x * 64;
    const float* yg = Y + ((size_t)b << 20);

    const int kk  = tid >> 4;
    const int nn4 = (tid & 15) << 2;
#pragma unroll
    for (int p = 0; p < 4; ++p) {
        const int k = p * 16 + kk;
        const float4 f = *reinterpret_cast<const float4*>(
            yg + (size_t)(k0 + k) * NDIM + n0 + nn4);
        t[(nn4 + 0) * 80 + k] = (uint8_t)(int)f.x;
        t[(nn4 + 1) * 80 + k] = (uint8_t)(int)f.y;
        t[(nn4 + 2) * 80 + k] = (uint8_t)(int)f.z;
        t[(nn4 + 3) * 80 + k] = (uint8_t)(int)f.w;
    }
    __syncthreads();

    const int n  = tid >> 2;
    const int kc = (tid & 3) << 4;
    const uint4 q = *reinterpret_cast<const uint4*>(t + n * 80 + kc);
    *reinterpret_cast<uint4*>(Ytq + ((size_t)(b * 1024 + n0 + n) << 10) + k0 + kc) = q;

    int s = 0;
    const uint32_t dw[4] = {q.x, q.y, q.z, q.w};
#pragma unroll
    for (int j = 0; j < 4; ++j) {
        const uint32_t d = dw[j];
        s += (int)((d & 255) + ((d >> 8) & 255) + ((d >> 16) & 255) + ((d >> 24) & 255));
    }
    atomicAdd(&Cs[b * 1024 + n0 + n], s);
}

__global__ __launch_bounds__(256) void qgemm_i8_pre_kernel(
    const uint8_t* __restrict__ Aq,
    const uint8_t* __restrict__ Btq,
    const int* __restrict__ Rs,
    const int* __restrict__ Cs,
    float* __restrict__ Out)
{
    __shared__ __attribute__((aligned(16))) uint8_t smA[128 * 80];
    __shared__ __attribute__((aligned(16))) uint8_t smB[128 * 80];

    const int tid = threadIdx.x;
    const int bz  = blockIdx.z;
    const int b   = bz & 7;
    const int m0  = blockIdx.y * 128;
    const int n0  = blockIdx.x * 128;

    const uint8_t* ag = Aq  + ((size_t)bz << 20);
    const uint8_t* bg = Btq + ((size_t)b  << 20);
    float*         og = Out + ((size_t)bz << 20);

    const int cp_r  = tid >> 2;
    const int cp_kc = (tid & 3) << 4;

    const int wave = tid >> 5;
    const int lane = tid & 31;
    const int c    = lane >> 4;
    const int r    = lane & 15;
    const int wm   = (wave & 3) * 32;
    const int wn   = (wave >> 2) * 64;

    v8i acc[2][4];
#pragma unroll
    for (int tm = 0; tm < 2; ++tm)
#pragma unroll
        for (int tn = 0; tn < 4; ++tn)
#pragma unroll
            for (int i = 0; i < 8; ++i) acc[tm][tn][i] = 0;

    for (int k0 = 0; k0 < KDIM; k0 += 64) {
        __syncthreads();

#pragma unroll
        for (int p = 0; p < 2; ++p) {
            const int rr = cp_r + p * 64;
            copy16_g2l(ag + (size_t)(m0 + rr) * KDIM + k0 + cp_kc, smA + rr * 80 + cp_kc);
            copy16_g2l(bg + (size_t)(n0 + rr) * KDIM + k0 + cp_kc, smB + rr * 80 + cp_kc);
        }
#ifdef HAVE_ASYNC_LDS
        async_wait_all();
#endif
        __syncthreads();

#pragma unroll
        for (int tm = 0; tm < 2; ++tm) {
            v8i afrag;
            const int am = wm + tm * 16 + r;
#pragma unroll
            for (int pp = 0; pp < 4; ++pp) {
                const uint2 d = *reinterpret_cast<const uint2*>(
                    smA + am * 80 + pp * 16 + c * 8);
                afrag[2 * pp]     = (int)d.x;
                afrag[2 * pp + 1] = (int)d.y;
            }
#pragma unroll
            for (int tn = 0; tn < 4; ++tn) {
                v8i bfrag;
                const int bn = wn + tn * 16 + r;
#pragma unroll
                for (int h = 0; h < 2; ++h) {
                    const uint2 d0 = *reinterpret_cast<const uint2*>(
                        smB + bn * 80 + h * 32 + c * 8);
                    const uint2 d1 = *reinterpret_cast<const uint2*>(
                        smB + bn * 80 + h * 32 + c * 8 + 16);
                    bfrag[4 * h]     = (int)d0.x;
                    bfrag[4 * h + 1] = (int)d0.y;
                    bfrag[4 * h + 2] = (int)d1.x;
                    bfrag[4 * h + 3] = (int)d1.y;
                }
                acc[tm][tn] = __builtin_amdgcn_wmma_i32_16x16x64_iu8(
                    true, afrag, false, bfrag, acc[tm][tn], false, false);
            }
        }
    }

    const float S   = 0.03f * 0.025f;
    const int   ZPC = KDIM * (-66) * 160;
    const int* rs_base = Rs + (size_t)bz * 1024 + m0;
    const int* cs_base = Cs + (size_t)b  * 1024 + n0;

#pragma unroll
    for (int tm = 0; tm < 2; ++tm) {
#pragma unroll
        for (int tn = 0; tn < 4; ++tn) {
            const int nloc = wn + tn * 16 + r;
            const int cs   = cs_base[nloc];
#pragma unroll
            for (int i = 0; i < 8; ++i) {
                const int mloc = wm + tm * 16 + i + 8 * c;
                const int corr = acc[tm][tn][i] - 160 * rs_base[mloc] + 66 * cs + ZPC;
                og[(size_t)(m0 + mloc) * NDIM + n0 + nloc] = S * (float)corr;
            }
        }
    }
}

__global__ __launch_bounds__(256) void qgemm_fused_kernel(
    const float* __restrict__ X, const float* __restrict__ Y, float* __restrict__ Out)
{
    __shared__ __attribute__((aligned(16))) uint8_t smA[128 * 72];
    __shared__ __attribute__((aligned(16))) uint8_t smB[128 * 72];
    __shared__ int rowsum[128];
    __shared__ int colsum[128];

    const int tid = threadIdx.x;
    const int bz  = blockIdx.z;
    const int b   = bz & 7;
    const int m0  = blockIdx.y * 128;
    const int n0  = blockIdx.x * 128;

    const float* xg = X + ((size_t)bz << 20);
    const float* yg = Y + ((size_t)b << 20);
    float*       og = Out + ((size_t)bz << 20);

    if (tid < 128) rowsum[tid] = 0; else colsum[tid - 128] = 0;

    const int a_row = tid >> 4;
    const int a_kc  = (tid & 15) << 2;
    const int b_n   = tid & 127;
    const int b_kg  = tid >> 7;

    int rsum_part[8];
#pragma unroll
    for (int p = 0; p < 8; ++p) rsum_part[p] = 0;
    int csum_part = 0;

    const int wave = tid >> 5, lane = tid & 31;
    const int c = lane >> 4, r = lane & 15;
    const int wm = (wave & 3) * 32, wn = (wave >> 2) * 64;

    v8i acc[2][4];
#pragma unroll
    for (int tm = 0; tm < 2; ++tm)
#pragma unroll
        for (int tn = 0; tn < 4; ++tn)
#pragma unroll
            for (int i = 0; i < 8; ++i) acc[tm][tn][i] = 0;

    for (int k0 = 0; k0 < KDIM; k0 += 64) {
        __syncthreads();
#pragma unroll
        for (int p = 0; p < 8; ++p) {
            const int m = a_row + p * 16;
            const float4 f = *reinterpret_cast<const float4*>(
                xg + (size_t)(m0 + m) * KDIM + k0 + a_kc);
            const int i0 = (int)f.x, i1 = (int)f.y, i2 = (int)f.z, i3 = (int)f.w;
            rsum_part[p] += i0 + i1 + i2 + i3;
            *reinterpret_cast<uint32_t*>(smA + m * 72 + a_kc) =
                ((uint32_t)(i0 & 255))       | ((uint32_t)(i1 & 255) << 8) |
                ((uint32_t)(i2 & 255) << 16) | ((uint32_t)(i3 & 255) << 24);
        }
#pragma unroll
        for (int p = 0; p < 8; ++p) {
            const int k = p * 8 + b_kg * 4;
            const size_t base = (size_t)(k0 + k) * NDIM + n0 + b_n;
            const int v0 = (int)yg[base];
            const int v1 = (int)yg[base + NDIM];
            const int v2 = (int)yg[base + 2 * (size_t)NDIM];
            const int v3 = (int)yg[base + 3 * (size_t)NDIM];
            csum_part += v0 + v1 + v2 + v3;
            *reinterpret_cast<uint32_t*>(smB + b_n * 72 + k) =
                ((uint32_t)(v0 & 255))       | ((uint32_t)(v1 & 255) << 8) |
                ((uint32_t)(v2 & 255) << 16) | ((uint32_t)(v3 & 255) << 24);
        }
        __syncthreads();
#pragma unroll
        for (int tm = 0; tm < 2; ++tm) {
            v8i afrag;
            const int am = wm + tm * 16 + r;
#pragma unroll
            for (int pp = 0; pp < 4; ++pp) {
                const uint2 d = *reinterpret_cast<const uint2*>(
                    smA + am * 72 + pp * 16 + c * 8);
                afrag[2 * pp] = (int)d.x;
                afrag[2 * pp + 1] = (int)d.y;
            }
#pragma unroll
            for (int tn = 0; tn < 4; ++tn) {
                v8i bfrag;
                const int bn = wn + tn * 16 + r;
#pragma unroll
                for (int h = 0; h < 2; ++h) {
                    const uint2 d0 = *reinterpret_cast<const uint2*>(
                        smB + bn * 72 + h * 32 + c * 8);
                    const uint2 d1 = *reinterpret_cast<const uint2*>(
                        smB + bn * 72 + h * 32 + c * 8 + 16);
                    bfrag[4 * h] = (int)d0.x;
                    bfrag[4 * h + 1] = (int)d0.y;
                    bfrag[4 * h + 2] = (int)d1.x;
                    bfrag[4 * h + 3] = (int)d1.y;
                }
                acc[tm][tn] = __builtin_amdgcn_wmma_i32_16x16x64_iu8(
                    true, afrag, false, bfrag, acc[tm][tn], false, false);
            }
        }
    }

#pragma unroll
    for (int p = 0; p < 8; ++p)
        atomicAdd(&rowsum[a_row + p * 16], rsum_part[p]);
    atomicAdd(&colsum[b_n], csum_part);
    __syncthreads();

    const float S = 0.03f * 0.025f;
    const int ZPC = KDIM * (-66) * 160;
#pragma unroll
    for (int tm = 0; tm < 2; ++tm) {
#pragma unroll
        for (int tn = 0; tn < 4; ++tn) {
            const int nloc = wn + tn * 16 + r;
            const int cs = colsum[nloc];
#pragma unroll
            for (int i = 0; i < 8; ++i) {
                const int mloc = wm + tm * 16 + i + 8 * c;
                const int corr = acc[tm][tn][i] - 160 * rowsum[mloc] + 66 * cs + ZPC;
                og[(size_t)(m0 + mloc) * NDIM + n0 + nloc] = S * (float)corr;
            }
        }
    }
}

extern "C" void kernel_launch(void* const* d_in, const int* in_sizes, int n_in,
                              void* d_out, int out_size, void* d_ws, size_t ws_size,
                              hipStream_t stream) {
    (void)in_sizes; (void)n_in; (void)out_size;
    const float* x = (const float*)d_in[0];
    const float* y = (const float*)d_in[1];
    float* out = (float*)d_out;

    const size_t XQ_BYTES = (size_t)S_DIM * B_DIM * MDIM * KDIM;
    const size_t YT_BYTES = (size_t)B_DIM * KDIM * NDIM;
    const size_t RS_BYTES = (size_t)S_DIM * B_DIM * MDIM * 4;
    const size_t CS_BYTES = (size_t)B_DIM * NDIM * 4;
    const size_t need = XQ_BYTES + YT_BYTES + RS_BYTES + CS_BYTES;

    dim3 block(256);
    dim3 gemm_grid(NDIM / 128, MDIM / 128, S_DIM * B_DIM);

    if (d_ws != nullptr && ws_size >= need) {
        uint8_t* xq  = (uint8_t*)d_ws;
        uint8_t* ytq = xq + XQ_BYTES;
        int*     rs  = (int*)(xq + XQ_BYTES + YT_BYTES);
        int*     cs  = (int*)(xq + XQ_BYTES + YT_BYTES + RS_BYTES);

        (void)hipMemsetAsync(cs, 0, CS_BYTES, stream);
        conv_x_kernel<<<S_DIM * B_DIM * MDIM, block, 0, stream>>>(x, (uint32_t*)xq, rs);
        conv_y_kernel<<<dim3(NDIM / 64, KDIM / 64, B_DIM), block, 0, stream>>>(y, ytq, cs);
        qgemm_i8_pre_kernel<<<gemm_grid, block, 0, stream>>>(xq, ytq, rs, cs, out);
    } else {
        qgemm_fused_kernel<<<gemm_grid, block, 0, stream>>>(x, y, out);
    }
}
